// RNNModel_54863912239797
// MI455X (gfx1250) — matrix operand order repaired and checked by reading, not compiled, not run on hardware
//
#include <hip/hip_runtime.h>

// ---------------------------------------------------------------------------
// Ring-attractor RNN, model_type=3, on MI455X (gfx1250).
//   r_rec[0] = 50*h,  r_rec[s+1] = A @ r_rec[s],  A = (1-c)I + c*W, c = 0.05
// 60 GEMMs of [4096x4096] x [4096x256], all fp32-accurate via split-bf16
// (hi/lo) 3-product WMMA: A_hi*b_hi + A_hi*b_lo + A_lo*b_hi, fp32 accumulate.
//
// Workspace layout (assumes ws_size >= 72 MB):
//   [0)                 W_hi swizzled  : 4096*4096 bf16 (32 MB), A-frag order
//   [32MB)              W_lo swizzled  : 32 MB
//   [64MB)              rT_hi[0]       : 256*4096 bf16 (2 MB)   (r transposed)
//   [66MB)              rT_lo[0]       : 2 MB
//   [68MB)              rT_hi[1]       : 2 MB
//   [70MB)              rT_lo[1]       : 2 MB
// ---------------------------------------------------------------------------

#define NN    4096        // neurons (M and K of the GEMM)
#define NB    256         // batch   (N of the GEMM)
#define NSTEP 60
#define COEF  0.05f       // dt/tau
#define KB    (NN / 32)   // 128 k-blocks of 32

typedef __attribute__((ext_vector_type(16))) __bf16 v16bf;
typedef __attribute__((ext_vector_type(8)))  __bf16 v8bf;
typedef __attribute__((ext_vector_type(8)))  float  v8f;

static __device__ __forceinline__ unsigned short f2bf_bits(float x) {
    __bf16 b = (__bf16)x;                       // RNE convert
    return __builtin_bit_cast(unsigned short, b);
}
static __device__ __forceinline__ float bf_bits2f(unsigned short u) {
    __bf16 b = __builtin_bit_cast(__bf16, u);
    return (float)b;
}

// ---------------------------------------------------------------------------
// Prep: A = COEF*W + (1-COEF)*I, split to bf16 hi/lo, store in a
// 16x32 bf16 A-fragment layout with contiguous k (not the ISA 7.12.2 table order):
//   lane L: m = L&15, half = L>>4; element e -> K = e + half*16
// Flat order: dword t = (((mt*KB + kb)*32 + lane)*8 + e2), e = 2*e2, 2*e2+1.
// One thread per packed dword -> 8,388,608 threads.
// ---------------------------------------------------------------------------
__global__ __launch_bounds__(256) void prep_w_kernel(
    const float* __restrict__ W,
    unsigned*    __restrict__ Whi,
    unsigned*    __restrict__ Wlo)
{
    const int t    = blockIdx.x * blockDim.x + threadIdx.x;
    const int e2   = t & 7;
    const int lane = (t >> 3) & 31;
    const int fk   = t >> 8;
    const int kb   = fk & (KB - 1);
    const int mt   = fk >> 7;
    const int m    = mt * 16 + (lane & 15);
    const int hsel = (lane >> 4) * 16;

    unsigned hp = 0, lp = 0;
#pragma unroll
    for (int j = 0; j < 2; ++j) {
        const int e = e2 * 2 + j;
        const int K = e + hsel;
        const int k = kb * 32 + K;
        float a = COEF * W[(size_t)m * NN + k] + (m == k ? (1.0f - COEF) : 0.0f);
        unsigned short h = f2bf_bits(a);
        unsigned short l = f2bf_bits(a - bf_bits2f(h));
        hp |= (unsigned)h << (16 * j);
        lp |= (unsigned)l << (16 * j);
    }
    Whi[t] = hp;
    Wlo[t] = lp;
}

// ---------------------------------------------------------------------------
// t_record
// ---------------------------------------------------------------------------
__global__ void t_record_kernel(float* __restrict__ t_rec) {
    int i = threadIdx.x;
    if (i <= NSTEP) t_rec[i] = 0.001f * (float)i;
}

// ---------------------------------------------------------------------------
// r0 = 50 * exp((cos(phi_i - theta_b) - 1) / kappa^2), two layouts.
// ---------------------------------------------------------------------------
__device__ __forceinline__ float r0_val(const float* theta, int i, int b) {
    const float PI = 3.14159265358979323846f;
    const float invk2 = 16.0f / (PI * PI);       // 1/kappa^2, kappa = pi/4
    float phi = (2.0f * PI / (float)NN) * (float)i;
    return 50.0f * expf((cosf(phi - theta[b]) - 1.0f) * invk2);
}

__global__ __launch_bounds__(256) void init_rec_kernel(
    const float* __restrict__ theta, float* __restrict__ rec0)
{
    int t = blockIdx.x * blockDim.x + threadIdx.x;   // b fastest: coalesced
    int b = t & (NB - 1);
    int i = t >> 8;
    rec0[t] = r0_val(theta, i, b);
}

__global__ __launch_bounds__(256) void init_rT_kernel(
    const float* __restrict__ theta,
    __bf16* __restrict__ rTh, __bf16* __restrict__ rTl)
{
    int t = blockIdx.x * blockDim.x + threadIdx.x;   // i fastest: coalesced rT
    int i = t & (NN - 1);
    int b = t >> 12;
    float x = r0_val(theta, i, b);
    unsigned short h = f2bf_bits(x);
    unsigned short l = f2bf_bits(x - bf_bits2f(h));
    rTh[(size_t)b * NN + i] = __builtin_bit_cast(__bf16, h);
    rTl[(size_t)b * NN + i] = __builtin_bit_cast(__bf16, l);
}

// ---------------------------------------------------------------------------
// One RNN step: rec_out = A @ r,  r given as transposed bf16 hi/lo.
// Block = 4 waves (128 thr), block tile 128M x 64N, wave tile 64M x 32N
// (4 x 2 tiles of 16x16). Per k-iter: 24 b128 loads, 24 v_wmma.
// ---------------------------------------------------------------------------
__global__ __launch_bounds__(128) void rnn_step_gemm(
    const __bf16* __restrict__ Whi, const __bf16* __restrict__ Wlo,
    const __bf16* __restrict__ rTh, const __bf16* __restrict__ rTl,
    float*  __restrict__ rec_out,
    __bf16* __restrict__ rTh_out, __bf16* __restrict__ rTl_out)
{
    const int lane = threadIdx.x & 31;
    const int wave = threadIdx.x >> 5;
    const int lid  = lane & 15;
    const int half = lane >> 4;

    const int m_base = blockIdx.x * 128 + (wave & 1) * 64;  // 4 M-tiles
    const int n_base = blockIdx.y * 64  + (wave >> 1) * 32; // 2 N-tiles
    const int mt0 = m_base >> 4;

    const v16bf* Ah = (const v16bf*)Whi;
    const v16bf* Al = (const v16bf*)Wlo;
    const v16bf* Bh = (const v16bf*)rTh;
    const v16bf* Bl = (const v16bf*)rTl;

    v8f acc[4][2];
#pragma unroll
    for (int mi = 0; mi < 4; ++mi)
#pragma unroll
        for (int ni = 0; ni < 2; ++ni) acc[mi][ni] = (v8f){};

    for (int kb = 0; kb < KB; ++kb) {
        v16bf a_hi[4], a_lo[4], b_hi[2], b_lo[2];
#pragma unroll
        for (int mi = 0; mi < 4; ++mi) {
            const size_t idx = ((size_t)(mt0 + mi) * KB + kb) * 32 + lane;
            a_hi[mi] = Ah[idx];
            a_lo[mi] = Al[idx];
        }
#pragma unroll
        for (int ni = 0; ni < 2; ++ni) {
            const int n = n_base + ni * 16 + lid;
            const size_t idx = (size_t)n * (NN / 16) + kb * 2 + half;
            b_hi[ni] = Bh[idx];
            b_lo[ni] = Bl[idx];
        }
#pragma unroll
        for (int mi = 0; mi < 4; ++mi) {
#pragma unroll
            for (int ni = 0; ni < 2; ++ni) {
                acc[mi][ni] = __builtin_amdgcn_wmma_f32_16x16x32_bf16(
                    false, a_hi[mi], false, b_hi[ni], (short)0, acc[mi][ni], false, false);
                acc[mi][ni] = __builtin_amdgcn_wmma_f32_16x16x32_bf16(
                    false, a_hi[mi], false, b_lo[ni], (short)0, acc[mi][ni], false, false);
                acc[mi][ni] = __builtin_amdgcn_wmma_f32_16x16x32_bf16(
                    false, a_lo[mi], false, b_hi[ni], (short)0, acc[mi][ni], false, false);
            }
        }
    }

    // Epilogue: fp32 record slice + transposed bf16 hi/lo for the next step.
    // C/D layout: VGPR v, lanes 0-15 -> M = v, lanes 16-31 -> M = v + 8.
#pragma unroll
    for (int mi = 0; mi < 4; ++mi) {
#pragma unroll
        for (int ni = 0; ni < 2; ++ni) {
            const int mrow0 = m_base + mi * 16 + half * 8;
            const int col   = n_base + ni * 16 + lid;
            v8bf vh, vl;
#pragma unroll
            for (int v = 0; v < 8; ++v) {
                float x = acc[mi][ni][v];
                rec_out[(size_t)(mrow0 + v) * NB + col] = x;
                unsigned short h = f2bf_bits(x);
                unsigned short l = f2bf_bits(x - bf_bits2f(h));
                vh[v] = __builtin_bit_cast(__bf16, h);
                vl[v] = __builtin_bit_cast(__bf16, l);
            }
            *(v8bf*)(rTh_out + (size_t)col * NN + mrow0) = vh;  // 16B aligned
            *(v8bf*)(rTl_out + (size_t)col * NN + mrow0) = vl;
        }
    }
}

// ---------------------------------------------------------------------------
extern "C" void kernel_launch(void* const* d_in, const int* in_sizes, int n_in,
                              void* d_out, int out_size, void* d_ws, size_t ws_size,
                              hipStream_t stream) {
    const float* theta = (const float*)d_in[0];   // [256]
    const float* W     = (const float*)d_in[1];   // [4096*4096]

    float* out   = (float*)d_out;
    float* t_rec = out;                            // [61]
    float* r_rec = out + 61;                       // [61, 4096, 256]

    const size_t WELEMS = (size_t)NN * NN;         // bf16 elems per W matrix
    const size_t RELEMS = (size_t)NB * NN;         // bf16 elems per rT buffer
    __bf16* Whi = (__bf16*)d_ws;
    __bf16* Wlo = Whi + WELEMS;
    __bf16* rTh[2] = { Wlo + WELEMS,              Wlo + WELEMS + 2 * RELEMS };
    __bf16* rTl[2] = { Wlo + WELEMS + RELEMS,     Wlo + WELEMS + 3 * RELEMS };

    // 1) one-time: A = c*W + (1-c)I, split-bf16, swizzled to A-fragment order
    prep_w_kernel<<<(WELEMS / 2) / 256, 256, 0, stream>>>((const float*)W,
                                                          (unsigned*)Whi,
                                                          (unsigned*)Wlo);
    // 2) t_record
    t_record_kernel<<<1, 64, 0, stream>>>(t_rec);

    // 3) r0 = 50*h in both layouts
    init_rec_kernel<<<((size_t)NN * NB) / 256, 256, 0, stream>>>(theta, r_rec);
    init_rT_kernel <<<((size_t)NN * NB) / 256, 256, 0, stream>>>(theta, rTh[0], rTl[0]);

    // 4) 60 GEMM steps, ping-ponging the transposed bf16 state
    dim3 grid(NN / 128, NB / 64);
    for (int s = 0; s < NSTEP; ++s) {
        const int cur = s & 1, nxt = cur ^ 1;
        rnn_step_gemm<<<grid, 128, 0, stream>>>(
            Whi, Wlo, rTh[cur], rTl[cur],
            r_rec + (size_t)(s + 1) * NN * NB,
            rTh[nxt], rTl[nxt]);
    }
}
